// Lin_Sfm_Attention_42949672960409
// MI455X (gfx1250) — matrix operand order repaired and checked by reading, not compiled, not run on hardware
//
#include <hip/hip_runtime.h>
#include <hip/hip_bf16.h>

// Problem constants (match reference)
constexpr int SEQ   = 2048;
constexpr int VOCAB = 2048;
constexpr int BATCH = 8;
constexpr float RSQRT2048 = 0.022097086912079608f;  // 1/sqrt(2048)

// ---------------- WMMA types (CDNA5 gfx1250, wave32) ----------------
typedef __attribute__((ext_vector_type(16))) _Float16     v16h;
typedef __attribute__((ext_vector_type(8)))  float        v8f;
typedef __attribute__((ext_vector_type(4)))  unsigned int v4u;
typedef __attribute__((ext_vector_type(4)))  int          v4i;

union Frag16 { v16h h; v4u q[2]; };

// ---------------- gfx1250 async global->LDS copies ----------------
#if __has_builtin(__builtin_amdgcn_global_load_async_to_lds_b128)
#define HAVE_ASYNC_LDS 1
typedef __attribute__((address_space(1))) v4i gv4i;   // global v4i
typedef __attribute__((address_space(3))) v4i lv4i;   // LDS v4i
template <int OFF>
__device__ __forceinline__ void async_b128(const void* g, void* l) {
  // LDS byte offset = low 32 bits of the generic LDS address (aperture in high bits)
  gv4i* gp = (gv4i*)(unsigned long long)(uintptr_t)g;
  lv4i* lp = (lv4i*)(unsigned int)(uintptr_t)l;
  __builtin_amdgcn_global_load_async_to_lds_b128(gp, lp, OFF, 0);
}
#endif

__device__ __forceinline__ void wait_async0() {
#if __has_builtin(__builtin_amdgcn_s_wait_asynccnt)
  __builtin_amdgcn_s_wait_asynccnt(0);
#else
  asm volatile("s_wait_asynccnt 0x0" ::: "memory");
#endif
}

__device__ inline float waveReduceSum(float v) {
  #pragma unroll
  for (int off = 16; off >= 1; off >>= 1) v += __shfl_xor(v, off, 32);
  return v;
}

// ---------------- f32 -> f16 conversion ----------------
__global__ __launch_bounds__(256)
void cvt_f32_to_f16(const float* __restrict__ src, _Float16* __restrict__ dst, int n) {
  int i = (blockIdx.x * 256 + threadIdx.x) * 4;
  if (i + 3 < n) {
    float4 v = *(const float4*)(src + i);
    dst[i + 0] = (_Float16)v.x;
    dst[i + 1] = (_Float16)v.y;
    dst[i + 2] = (_Float16)v.z;
    dst[i + 3] = (_Float16)v.w;
  }
}

// ---------------- WMMA NT GEMM: C[m,n] = sum_k A[m,k]*B[n,k] ----------------
// A,B: row-major f16, ld = 2048. Workgroup computes a 128x128 tile.
// 8 waves arranged 2(M) x 4(N); each wave: 64x32 = 4x2 fragments of 16x16.
// K staged 64-wide through double-buffered LDS via async global->LDS loads:
// one barrier per slab; next slab's loads overlap the current 16 WMMAs.
// If Cf != null: f32 out scaled by fixed_scale*(*scale_ptr).  Else f16 out.
#define LDH 72   // LDS row stride in halfs (144 B, multiple of 16 B)

__global__ __launch_bounds__(256)
void wmma_gemm_nt(const _Float16* __restrict__ A,
                  const _Float16* __restrict__ B,
                  _Float16* __restrict__ Ch,
                  float*     __restrict__ Cf,
                  const float* __restrict__ scale_ptr,
                  float fixed_scale,
                  int causal_skip)
{
  const int bx = blockIdx.x;   // N tile (t / column block)
  const int by = blockIdx.y;   // M tile (s / row block)
  if (causal_skip && bx > by) return;   // entire tile strictly above diagonal

  __shared__ alignas(16) _Float16 lA[2][128 * LDH];
  __shared__ alignas(16) _Float16 lB[2][128 * LDH];

  const int tid  = threadIdx.x;
  const int lane = tid & 31;
  const int wv   = tid >> 5;
  const int wm   = wv >> 2;     // 0..1
  const int wn   = wv & 3;      // 0..3
  const int kh   = lane >> 4;   // lane-half selector (ISA 16-bit layouts)
  const int l16  = lane & 15;

  const int m0 = by * 128;
  const int n0 = bx * 128;

  v8f acc[4][2];
  #pragma unroll
  for (int i = 0; i < 4; i++)
    #pragma unroll
    for (int j = 0; j < 2; j++)
      #pragma unroll
      for (int r = 0; r < 8; r++) acc[i][j][r] = 0.0f;

  const int crow  = tid >> 1;   // 0..127 : row copied by this thread
  const int chalf = tid & 1;    // which 32-half of the 64-wide K slab

  // stage slab starting at k0 into buffer p (each thread: 64 B of A + 64 B of B)
  auto stage = [&](int k0, int p) {
    const _Float16* ga = A + (size_t)(m0 + crow) * 2048 + k0 + chalf * 32;
    const _Float16* gb = B + (size_t)(n0 + crow) * 2048 + k0 + chalf * 32;
    _Float16* sa = &lA[p][crow * LDH + chalf * 32];
    _Float16* sb = &lB[p][crow * LDH + chalf * 32];
#ifdef HAVE_ASYNC_LDS
    async_b128<0>(ga, sa);  async_b128<16>(ga, sa);
    async_b128<32>(ga, sa); async_b128<48>(ga, sa);
    async_b128<0>(gb, sb);  async_b128<16>(gb, sb);
    async_b128<32>(gb, sb); async_b128<48>(gb, sb);
#else
    const v4u* g4a = (const v4u*)ga;  const v4u* g4b = (const v4u*)gb;
    v4u* s4a = (v4u*)sa;              v4u* s4b = (v4u*)sb;
    #pragma unroll
    for (int q = 0; q < 4; q++) { s4a[q] = g4a[q]; s4b[q] = g4b[q]; }
#endif
  };

  stage(0, 0);
  int p = 0;
  for (int k0 = 0; k0 < 2048; k0 += 64) {
#ifdef HAVE_ASYNC_LDS
    wait_async0();            // this wave's async stores to LDS done
#endif
    __syncthreads();          // all waves' slab-p data visible

    if (k0 + 64 < 2048) stage(k0 + 64, p ^ 1);   // overlap with compute below

    #pragma unroll
    for (int ks = 0; ks < 64; ks += 32) {
      // B fragments (32x16, K x N): lanes 0-15 -> N=lane, K=0..15; lanes 16-31 -> K=16..31.
      Frag16 bf[2];
      #pragma unroll
      for (int fn = 0; fn < 2; fn++) {
        const _Float16* pb = &lB[p][(wn * 32 + fn * 16 + l16) * LDH + ks + kh * 16];
        bf[fn].q[0] = *(const v4u*)(pb);
        bf[fn].q[1] = *(const v4u*)(pb + 8);
      }
      // A fragments (16x32, M x K): elems 0-15 -> K=16kh..16kh+15
      #pragma unroll
      for (int fm = 0; fm < 4; fm++) {
        Frag16 af;
        const _Float16* pa = &lA[p][(wm * 64 + fm * 16 + l16) * LDH + ks + kh * 16];
        af.q[0] = *(const v4u*)(pa);
        af.q[1] = *(const v4u*)(pa + 8);
        #pragma unroll
        for (int fn = 0; fn < 2; fn++) {
          acc[fm][fn] = __builtin_amdgcn_wmma_f32_16x16x32_f16(
              false, af.h, false, bf[fn].h, (short)0, acc[fm][fn], false, false);
        }
      }
    }
    p ^= 1;
  }

  float sc = fixed_scale;
  if (scale_ptr) sc *= scale_ptr[0];

  // C layout: elem r -> M = 8*(lane>>4)+r, N = lane&15
  #pragma unroll
  for (int fm = 0; fm < 4; fm++) {
    #pragma unroll
    for (int fn = 0; fn < 2; fn++) {
      const int Mb = m0 + wm * 64 + fm * 16 + kh * 8;
      const int Nb = n0 + wn * 32 + fn * 16 + l16;
      #pragma unroll
      for (int r = 0; r < 8; r++) {
        float v = acc[fm][fn][r] * sc;
        size_t off = (size_t)(Mb + r) * 2048 + Nb;
        if (Cf) Cf[off] = v;
        else    Ch[off] = (_Float16)v;
      }
    }
  }
}

// ---------------- causal row softmax, in place; row 0 zeroed ----------------
__global__ __launch_bounds__(256)
void softmax_causal_rows(float* __restrict__ S1) {
  const int s   = blockIdx.x;
  const int tid = threadIdx.x;
  float* row = S1 + (size_t)s * SEQ;
  __shared__ float red[256];
  if (s == 0) {                       // reference zeroes A1[0,:] after softmax
    for (int t = tid; t < SEQ; t += 256) row[t] = 0.f;
    return;
  }
  const int n = s + 1;                // valid causal length
  float m = -3.0e38f;
  for (int t = tid; t < n; t += 256) m = fmaxf(m, row[t]);
  red[tid] = m; __syncthreads();
  for (int o = 128; o > 0; o >>= 1) { if (tid < o) red[tid] = fmaxf(red[tid], red[tid + o]); __syncthreads(); }
  m = red[0]; __syncthreads();
  float l = 0.f;
  for (int t = tid; t < n; t += 256) l += __expf(row[t] - m);
  red[tid] = l; __syncthreads();
  for (int o = 128; o > 0; o >>= 1) { if (tid < o) red[tid] += red[tid + o]; __syncthreads(); }
  const float inv = 1.0f / red[0];
  for (int t = tid; t < SEQ; t += 256)
    row[t] = (t < n) ? __expf(row[t] - m) * inv : 0.f;
}

// ------- out[b,v] = scale * sum_t w[b_or_shared][t] * emb[idx[b,t], v] -------
__global__ __launch_bounds__(128)
void emb_weighted_sum(const float* __restrict__ wvec, int wstride,
                      const int* __restrict__ idx,
                      const float* __restrict__ emb,
                      float* __restrict__ out,
                      const float* __restrict__ scale_ptr)
{
  const int b  = blockIdx.y;
  const int v0 = blockIdx.x * 512 + threadIdx.x * 4;
  const float* wrow = wvec + (size_t)b * wstride;
  const int*   irow = idx + (size_t)b * SEQ;
  float4 acc = {0.f, 0.f, 0.f, 0.f};
  for (int t = 0; t < SEQ; ++t) {
    const float a = wrow[t];
    const float4 e = *(const float4*)(emb + (size_t)irow[t] * VOCAB + v0);
    acc.x += a * e.x; acc.y += a * e.y; acc.z += a * e.z; acc.w += a * e.w;
  }
  const float sc = scale_ptr ? scale_ptr[0] : 1.0f;
  float4 r = {acc.x * sc, acc.y * sc, acc.z * sc, acc.w * sc};
  *(float4*)(out + (size_t)b * VOCAB + v0) = r;
}

// ------- out[b,o] = sum_v M[o,v]*x[b,v] (+ emb[idx[b,S-1],o] if idx) -------
__global__ __launch_bounds__(256)
void gemv_row(const float* __restrict__ M, const float* __restrict__ x,
              float* __restrict__ out,
              const int* __restrict__ idx, const float* __restrict__ emb)
{
  const int b    = blockIdx.y;
  const int lane = threadIdx.x & 31;
  const int o    = blockIdx.x * 8 + (threadIdx.x >> 5);
  const float4* mr = (const float4*)(M + (size_t)o * VOCAB);
  const float4* xr = (const float4*)(x + (size_t)b * VOCAB);
  float acc = 0.f;
  for (int i = lane; i < VOCAB / 4; i += 32) {
    float4 a = mr[i], c = xr[i];
    acc += a.x * c.x + a.y * c.y + a.z * c.z + a.w * c.w;
  }
  acc = waveReduceSum(acc);
  if (lane == 0) {
    float add = 0.f;
    if (idx) add = emb[(size_t)idx[b * SEQ + SEQ - 1] * VOCAB + o];
    out[(size_t)b * VOCAB + o] = acc + add;
  }
}

// ------- out[b,v] = sum_o M[o,v]*x[b,o]  (M^T gemv, coalesced over v) -------
__global__ __launch_bounds__(256)
void gemv_col(const float* __restrict__ M, const float* __restrict__ x,
              float* __restrict__ out)
{
  const int b = blockIdx.y;
  const int v = blockIdx.x * 256 + threadIdx.x;
  const float* xb = x + (size_t)b * VOCAB;
  float acc = 0.f;
  for (int o = 0; o < VOCAB; ++o)
    acc += M[(size_t)o * VOCAB + v] * xb[o];
  out[(size_t)b * VOCAB + v] = acc;
}

// ------- er[b,s]=E[b,s]·r[b], ew[b,s]=E[b,s]·w[b]; one wave per (b,s) -------
__global__ __launch_bounds__(256)
void er_ew_kernel(const int* __restrict__ idx, const float* __restrict__ emb,
                  const float* __restrict__ r, const float* __restrict__ w,
                  float* __restrict__ er, float* __restrict__ ew)
{
  const int gw   = blockIdx.x * 8 + (threadIdx.x >> 5);
  const int lane = threadIdx.x & 31;
  const int b = gw >> 11;        // SEQ = 2^11
  const int s = gw & (SEQ - 1);
  const float4* e4 = (const float4*)(emb + (size_t)idx[b * SEQ + s] * VOCAB);
  const float4* r4 = (const float4*)(r + (size_t)b * VOCAB);
  const float4* w4 = (const float4*)(w + (size_t)b * VOCAB);
  float ar = 0.f, aw_ = 0.f;
  for (int i = lane; i < VOCAB / 4; i += 32) {
    float4 e = e4[i], rr = r4[i], ww = w4[i];
    ar  += e.x * rr.x + e.y * rr.y + e.z * rr.z + e.w * rr.w;
    aw_ += e.x * ww.x + e.y * ww.y + e.z * ww.z + e.w * ww.w;
  }
  ar  = waveReduceSum(ar);
  aw_ = waveReduceSum(aw_);
  if (lane == 0) { er[b * SEQ + s] = ar; ew[b * SEQ + s] = aw_; }
}

// ------- aw[b,s] = sum_t A1[s,t]*ew[b,t]; one wave per s, 8 batch accs -------
__global__ __launch_bounds__(256)
void aw_kernel(const float* __restrict__ A1, const float* __restrict__ ew,
               float* __restrict__ aw)
{
  const int s    = blockIdx.x * 8 + (threadIdx.x >> 5);
  const int lane = threadIdx.x & 31;
  const float* row = A1 + (size_t)s * SEQ;
  float acc[BATCH];
  #pragma unroll
  for (int b = 0; b < BATCH; b++) acc[b] = 0.f;
  for (int t = lane; t < SEQ; t += 32) {
    const float a = row[t];            // zeros above diagonal -> safe full row
    #pragma unroll
    for (int b = 0; b < BATCH; ++b) acc[b] += a * ew[b * SEQ + t];
  }
  #pragma unroll
  for (int b = 0; b < BATCH; ++b) {
    float v = waveReduceSum(acc[b]);
    if (lane == 0) aw[b * SEQ + s] = v;
  }
}

// ------- stage-2 softmax over s: A2[b,s] -------
__global__ __launch_bounds__(256)
void softmax2_kernel(const float* __restrict__ er, const float* __restrict__ aw,
                     const float* __restrict__ beta2, float* __restrict__ A2)
{
  const int b   = blockIdx.x;
  const int tid = threadIdx.x;
  __shared__ float red[256];
  const float sc = beta2[0] * RSQRT2048;
  float m = -3.0e38f;
  for (int s = tid; s < SEQ; s += 256) {
    float x;
    if (s >= SEQ - 2) x = -1e30f;                       // last_two mask
    else x = sc * (er[b * SEQ + s] + (s > 0 ? aw[b * SEQ + s] : 0.f));
    A2[b * SEQ + s] = x;
    m = fmaxf(m, x);
  }
  red[tid] = m; __syncthreads();
  for (int o = 128; o > 0; o >>= 1) { if (tid < o) red[tid] = fmaxf(red[tid], red[tid + o]); __syncthreads(); }
  m = red[0]; __syncthreads();
  float l = 0.f;
  for (int s = tid; s < SEQ; s += 256) l += __expf(A2[b * SEQ + s] - m);
  red[tid] = l; __syncthreads();
  for (int o = 128; o > 0; o >>= 1) { if (tid < o) red[tid] += red[tid + o]; __syncthreads(); }
  const float inv = 1.0f / red[0];
  for (int s = tid; s < SEQ; s += 256) A2[b * SEQ + s] = __expf(A2[b * SEQ + s] - m) * inv;
}

// =======================================================================
extern "C" void kernel_launch(void* const* d_in, const int* in_sizes, int n_in,
                              void* d_out, int out_size, void* d_ws, size_t ws_size,
                              hipStream_t stream)
{
  (void)in_sizes; (void)n_in; (void)out_size;
  const int*   idx   = (const int*)  d_in[0];
  const float* emb   = (const float*)d_in[1];
  const float* pos   = (const float*)d_in[2];
  const float* WQ1   = (const float*)d_in[3];
  const float* WK1   = (const float*)d_in[4];
  const float* WV1   = (const float*)d_in[5];
  const float* WQ2   = (const float*)d_in[6];
  const float* WK2   = (const float*)d_in[7];
  const float* beta1 = (const float*)d_in[8];
  const float* beta2 = (const float*)d_in[9];
  const float* betaO = (const float*)d_in[10];
  float* out = (float*)d_out;

  const size_t MB = 1024ull * 1024ull;
  if (ws_size < 41 * MB) return;   // need ~40.6 MB of scratch

  char* ws = (char*)d_ws;
  // f16 staging (reused: S1/A1 f32 [0,16MB) overwrites dead P16/WQ16)
  _Float16* P16  = (_Float16*)(ws +  0 * MB);
  _Float16* WQ16 = (_Float16*)(ws +  8 * MB);
  _Float16* WK16 = (_Float16*)(ws + 16 * MB);
  _Float16* Q16  = (_Float16*)(ws + 24 * MB);
  _Float16* K16  = (_Float16*)(ws + 32 * MB);
  float* S1 = (float*)(ws);                      // 16 MB; becomes A1 in place
  float* ylast = (float*)(ws + 40 * MB);
  float* zlast = ylast + BATCH * VOCAB;
  float* q2    = zlast + BATCH * VOCAB;
  float* rvec  = q2    + BATCH * VOCAB;
  float* wvec  = rvec  + BATCH * VOCAB;
  float* er    = wvec  + BATCH * VOCAB;
  float* ew    = er    + BATCH * SEQ;
  float* aw    = ew    + BATCH * SEQ;
  float* A2    = aw    + BATCH * SEQ;

  // 1) f32 -> f16 for the WMMA GEMM inputs
  const int NCVT = (SEQ * SEQ) / (256 * 4);
  cvt_f32_to_f16<<<NCVT, 256, 0, stream>>>(pos, P16, SEQ * SEQ);
  cvt_f32_to_f16<<<NCVT, 256, 0, stream>>>(WQ1, WQ16, SEQ * SEQ);
  cvt_f32_to_f16<<<NCVT, 256, 0, stream>>>(WK1, WK16, SEQ * SEQ);

  // 2) Q1 = P*WQ1^T, K1 = P*WK1^T (f16 out); S1 = beta1/sqrt(S) * Q1*K1^T (f32, causal)
  dim3 ggrid(16, 16);
  wmma_gemm_nt<<<ggrid, 256, 0, stream>>>(P16, WQ16, Q16, nullptr, nullptr, 1.0f, 0);
  wmma_gemm_nt<<<ggrid, 256, 0, stream>>>(P16, WK16, K16, nullptr, nullptr, 1.0f, 0);
  wmma_gemm_nt<<<ggrid, 256, 0, stream>>>(Q16, K16, nullptr, S1, beta1, RSQRT2048, 1);

  // 3) A1 = causal softmax rows of S1, in place; row 0 zeroed
  softmax_causal_rows<<<SEQ, 256, 0, stream>>>(S1);

  // 4) ylast[b] = sum_t A1[S-1,t] * emb[idx[b,t]]
  emb_weighted_sum<<<dim3(VOCAB / 512, BATCH), 128, 0, stream>>>(
      S1 + (size_t)(SEQ - 1) * SEQ, 0, idx, emb, ylast, nullptr);

  // 5) GEMV chain: zlast = E[b,-1] + WV1*ylast ; q2 = WQ2*zlast ; r = WK2^T*q2 ; w = WV1^T*r
  gemv_row<<<dim3(VOCAB / 8, BATCH), 256, 0, stream>>>(WV1, ylast, zlast, idx, emb);
  gemv_row<<<dim3(VOCAB / 8, BATCH), 256, 0, stream>>>(WQ2, zlast, q2, nullptr, nullptr);
  gemv_col<<<dim3(VOCAB / 256, BATCH), 256, 0, stream>>>(WK2, q2, rvec);
  gemv_col<<<dim3(VOCAB / 256, BATCH), 256, 0, stream>>>(WV1, rvec, wvec);

  // 6) er/ew dots, aw = A1*ew, stage-2 softmax, final weighted gather
  er_ew_kernel<<<(BATCH * SEQ) / 8, 256, 0, stream>>>(idx, emb, rvec, wvec, er, ew);
  aw_kernel<<<SEQ / 8, 256, 0, stream>>>(S1, ew, aw);
  softmax2_kernel<<<BATCH, 256, 0, stream>>>(er, aw, beta2, A2);
  emb_weighted_sum<<<dim3(VOCAB / 512, BATCH), 128, 0, stream>>>(
      A2, SEQ, idx, emb, out, betaO);
}
